// MoEFFN_28535762715052
// MI455X (gfx1250) — compile-verified, matrix operand order checked by reading, not run on hardware
//
#include <hip/hip_runtime.h>
#include <hip/hip_bf16.h>
#include <math.h>

// ---------------------------------------------------------------------------
// MoE FFN for MI455X (gfx1250, wave32, WMMA bf16 16x16x32)
//
// Roofline: ~155 GFLOP, ~0.9 GB weight traffic -> compute-leaning at
// 23.3 TB/s, so all matmuls go through v_wmma_f32_16x16x32_bf16 (f32
// accumulate). Weights stay f32 in memory and are converted per-tile with
// v_cvt_pk_bf16_f32: a per-launch bf16 pre-conversion pass would add ~900 MB
// of extra HBM traffic (~39 us) and cannot be cached across calls.
// Each wave computes NSUB n-subtiles sharing one A fragment to amortize
// activation loads and addressing across multiple WMMAs.
// ---------------------------------------------------------------------------
constexpr int N_TOK = 2048;   // B*T
constexpr int DIM   = 2048;   // D
constexpr int NEXP  = 64;     // E
constexpr int TOPK  = 4;      // K
constexpr int HDIM  = 512;    // H
constexpr int NSH   = 2;      // S
constexpr int HSH   = 2048;   // H*K shared hidden
constexpr int CAP   = 512;    // per-expert capacity

typedef __bf16 bf16_t;
typedef __attribute__((ext_vector_type(16))) __bf16 v16bf;
typedef __attribute__((ext_vector_type(8)))  float  v8f;

__device__ __forceinline__ v8f wmma_bf16(v16bf a, v16bf b, v8f c) {
  // D = A(16x32 bf16) x B(32x16 bf16) + C(16x16 f32)
  return __builtin_amdgcn_wmma_f32_16x16x32_bf16(
      /*neg_a=*/false, a, /*neg_b=*/false, b,
      /*c_mod=*/(short)0, c, /*reuse_a=*/false, /*reuse_b=*/false);
}

// 16 contiguous f32 -> one lane's bf16 A-fragment half (4x b128 + pk cvt).
__device__ __forceinline__ v16bf load16_cvt(const float* __restrict__ p) {
  const float4 f0 = ((const float4*)p)[0];
  const float4 f1 = ((const float4*)p)[1];
  const float4 f2 = ((const float4*)p)[2];
  const float4 f3 = ((const float4*)p)[3];
  v16bf a;
  a[0]=(bf16_t)f0.x;  a[1]=(bf16_t)f0.y;  a[2]=(bf16_t)f0.z;  a[3]=(bf16_t)f0.w;
  a[4]=(bf16_t)f1.x;  a[5]=(bf16_t)f1.y;  a[6]=(bf16_t)f1.z;  a[7]=(bf16_t)f1.w;
  a[8]=(bf16_t)f2.x;  a[9]=(bf16_t)f2.y;  a[10]=(bf16_t)f2.z; a[11]=(bf16_t)f2.w;
  a[12]=(bf16_t)f3.x; a[13]=(bf16_t)f3.y; a[14]=(bf16_t)f3.z; a[15]=(bf16_t)f3.w;
  return a;
}

// 16 strided f32 (weight column) -> one lane's bf16 B-fragment half.
__device__ __forceinline__ v16bf load16_cvt_strided(const float* __restrict__ p,
                                                    size_t stride) {
  v16bf b;
#pragma unroll
  for (int j = 0; j < 16; ++j) b[j] = (bf16_t)p[(size_t)j * stride];
  return b;
}

__device__ __forceinline__ float silu(float g) {
  return g * (1.f / (1.f + __expf(-g)));
}

// ---------------------------------------------------------------------------
// 0) zero the output (harness poisons d_out)
// ---------------------------------------------------------------------------
__global__ void moe_zero_out(float* __restrict__ out) {
  size_t i = (size_t)blockIdx.x * blockDim.x + threadIdx.x;
  ((float4*)out)[i] = make_float4(0.f, 0.f, 0.f, 0.f);
}

// ---------------------------------------------------------------------------
// 1) router: one wave per token; 64 experts -> 2 logits/lane; wave-shuffle
//    softmax + iterative top-4 argmax + renormalize.
// ---------------------------------------------------------------------------
__global__ void moe_router(const float* __restrict__ x,
                           const float* __restrict__ Wr,
                           const float* __restrict__ bias,
                           int* __restrict__ eid,
                           float* __restrict__ wout) {
  int tok  = (int)((blockIdx.x * blockDim.x + threadIdx.x) >> 5);
  int lane = threadIdx.x & 31;
  if (tok >= N_TOK) return;
  const float* xr = x + (size_t)tok * DIM;

  float l0 = 0.f, l1 = 0.f;
  for (int d = 0; d < DIM; d += 4) {
    float4 xv = *(const float4*)(xr + d);
    const float* wr = Wr + (size_t)d * NEXP + lane;
    l0 += xv.x * wr[0]         + xv.y * wr[NEXP]
        + xv.z * wr[2*NEXP]    + xv.w * wr[3*NEXP];
    l1 += xv.x * wr[32]        + xv.y * wr[NEXP+32]
        + xv.z * wr[2*NEXP+32] + xv.w * wr[3*NEXP+32];
  }
  l0 += bias[lane];
  l1 += bias[lane + 32];

  float m = fmaxf(l0, l1);
  for (int off = 16; off; off >>= 1) m = fmaxf(m, __shfl_xor(m, off, 32));
  float p0 = __expf(l0 - m), p1 = __expf(l1 - m);
  float s = p0 + p1;
  for (int off = 16; off; off >>= 1) s += __shfl_xor(s, off, 32);
  p0 /= s; p1 /= s;

  int   cid[TOPK];
  float cs[TOPK];
#pragma unroll
  for (int r = 0; r < TOPK; ++r) {
    float v = p0; int id = lane;
    if (p1 > v) { v = p1; id = lane + 32; }
    for (int off = 16; off; off >>= 1) {
      float ov = __shfl_xor(v, off, 32);
      int  oid = __shfl_xor(id, off, 32);
      if (ov > v || (ov == v && oid < id)) { v = ov; id = oid; }
    }
    cid[r] = id; cs[r] = v;
    if (id == lane)      p0 = -1.f;
    if (id == lane + 32) p1 = -1.f;
  }
  float tot = cs[0] + cs[1] + cs[2] + cs[3];
  if (lane < TOPK) {
    eid[tok * TOPK + lane]  = cid[lane];
    wout[tok * TOPK + lane] = cs[lane] / tot;
  }
}

// ---------------------------------------------------------------------------
// 2) deterministic in-order capacity assignment (reference cumsum semantics).
// ---------------------------------------------------------------------------
__global__ void moe_build_slots(const int* __restrict__ eid,
                                const float* __restrict__ w,
                                int* __restrict__ counts,
                                int* __restrict__ slot_tok,
                                float* __restrict__ slot_w) {
  __shared__ int cnt[NEXP];
  int t = threadIdx.x;
  if (t < NEXP) cnt[t] = 0;
  __syncthreads();
  if (t == 0) {
    for (int i = 0; i < N_TOK * TOPK; ++i) {
      int e = eid[i];
      int c = cnt[e]++;
      if (c < CAP) {
        slot_tok[e * CAP + c] = i / TOPK;
        slot_w[e * CAP + c]   = w[i];
      }
    }
  }
  __syncthreads();
  if (t < NEXP) counts[t] = min(cnt[t], CAP);
}

// ---------------------------------------------------------------------------
// 3) expert GEMM1: h = silu(x_g@Wg) * (x_g@Wu). Gathered A rows via slot_tok.
//    Per wave: 16 rows x (NSUB1=2)x16 cols, one shared A fragment per k-step,
//    4 WMMAs per k-step. Tiles past the expert fill count exit early.
// ---------------------------------------------------------------------------
constexpr int NSUB1 = 2;   // n-subtiles per wave for the 2-matrix GEMMs
constexpr int NSUB2 = 4;   // n-subtiles per wave for the 1-matrix GEMMs

__global__ void moe_expert_gemm1(const float* __restrict__ x,
                                 const float* __restrict__ Wg,
                                 const float* __restrict__ Wu,
                                 const int* __restrict__ counts,
                                 const int* __restrict__ slot_tok,
                                 bf16_t* __restrict__ hbuf) {
  constexpr int MT = CAP / 16, NG = (HDIM / 16) / NSUB1;  // 32, 16
  int wid = blockIdx.x * (blockDim.x >> 5) + (threadIdx.x >> 5);
  int ng = wid % NG;
  int mt = (wid / NG) % MT;
  int e  = wid / (NG * MT);

  int count   = counts[e];
  int rowbase = mt * 16;
  if (rowbase >= count) return;     // block-uniform early exit

  int lane  = threadIdx.x & 31;
  int lrow  = lane & 15;
  int khalf = lane >> 4;

  int r   = rowbase + lrow;
  int tok = (r < count) ? slot_tok[e * CAP + r] : -1;
  const float* arow = x + (size_t)(tok < 0 ? 0 : tok) * DIM;

  const float* wgp[NSUB1];
  const float* wup[NSUB1];
#pragma unroll
  for (int s = 0; s < NSUB1; ++s) {
    int col = (ng * NSUB1 + s) * 16 + lrow;
    wgp[s] = Wg + (size_t)e * DIM * HDIM + col;
    wup[s] = Wu + (size_t)e * DIM * HDIM + col;
  }

  v8f accg[NSUB1] = {}, accu[NSUB1] = {};
  for (int k0 = 0; k0 < DIM; k0 += 32) {
    int ka = k0 + khalf * 16;
    v16bf a = {};
    if (tok >= 0) a = load16_cvt(arow + ka);
#pragma unroll
    for (int s = 0; s < NSUB1; ++s) {
      v16bf bg = load16_cvt_strided(wgp[s] + (size_t)ka * HDIM, HDIM);
      accg[s] = wmma_bf16(a, bg, accg[s]);
      v16bf bu = load16_cvt_strided(wup[s] + (size_t)ka * HDIM, HDIM);
      accu[s] = wmma_bf16(a, bu, accu[s]);
    }
  }

  // fused SwiGLU + bf16 store (C/D layout: lane->col, vgpr v -> row v+8*khalf)
#pragma unroll
  for (int v = 0; v < 8; ++v) {
    int row = rowbase + v + khalf * 8;
    if (row < count) {
      bf16_t* hr = hbuf + (size_t)(e * CAP + row) * HDIM;
#pragma unroll
      for (int s = 0; s < NSUB1; ++s) {
        int ncol = (ng * NSUB1 + s) * 16 + lrow;
        hr[ncol] = (bf16_t)(silu(accg[s][v]) * accu[s][v]);
      }
    }
  }
}

// ---------------------------------------------------------------------------
// 4) expert GEMM2 + combine: yb = h @ Wd; weighted atomic scatter into out.
//    Per wave: 16 rows x (NSUB2=4)x16 cols, shared A fragment per k-step.
// ---------------------------------------------------------------------------
__global__ void moe_expert_gemm2(const bf16_t* __restrict__ hbuf,
                                 const float* __restrict__ Wd,
                                 const int* __restrict__ counts,
                                 const int* __restrict__ slot_tok,
                                 const float* __restrict__ slot_w,
                                 float* __restrict__ out) {
  constexpr int MT = CAP / 16, NG = (DIM / 16) / NSUB2;   // 32, 32
  int wid = blockIdx.x * (blockDim.x >> 5) + (threadIdx.x >> 5);
  int ng = wid % NG;
  int mt = (wid / NG) % MT;
  int e  = wid / (NG * MT);

  int count   = counts[e];
  int rowbase = mt * 16;
  if (rowbase >= count) return;

  int lane  = threadIdx.x & 31;
  int lrow  = lane & 15;
  int khalf = lane >> 4;

  const bf16_t* hrow = hbuf + (size_t)(e * CAP + rowbase + lrow) * HDIM;
  bool rvalid = (rowbase + lrow) < count;

  const float* wdp[NSUB2];
#pragma unroll
  for (int s = 0; s < NSUB2; ++s)
    wdp[s] = Wd + (size_t)e * HDIM * DIM + (ng * NSUB2 + s) * 16 + lrow;

  v8f acc[NSUB2] = {};
  for (int k0 = 0; k0 < HDIM; k0 += 32) {
    int ka = k0 + khalf * 16;
    v16bf a = {};
    if (rvalid) a = *(const v16bf*)(hrow + ka);  // bf16 contiguous, 32B aligned
#pragma unroll
    for (int s = 0; s < NSUB2; ++s) {
      v16bf b = load16_cvt_strided(wdp[s] + (size_t)ka * DIM, DIM);
      acc[s] = wmma_bf16(a, b, acc[s]);
    }
  }

#pragma unroll
  for (int v = 0; v < 8; ++v) {
    int row = rowbase + v + khalf * 8;
    if (row < count) {
      int   tok = slot_tok[e * CAP + row];
      float wgt = slot_w[e * CAP + row];
      float* orow = out + (size_t)tok * DIM;
#pragma unroll
      for (int s = 0; s < NSUB2; ++s) {
        int ncol = (ng * NSUB2 + s) * 16 + lrow;
        atomicAdd(orow + ncol, wgt * acc[s][v]);
      }
    }
  }
}

// ---------------------------------------------------------------------------
// 5) shared experts GEMM1: hs = silu(x@Sg) * (x@Su), bf16 staged.
// ---------------------------------------------------------------------------
__global__ void moe_shared_gemm1(const float* __restrict__ x,
                                 const float* __restrict__ Sg,
                                 const float* __restrict__ Su,
                                 bf16_t* __restrict__ hsbuf) {
  constexpr int MT = N_TOK / 16, NG = (HSH / 16) / NSUB1;  // 128, 64
  int wid = blockIdx.x * (blockDim.x >> 5) + (threadIdx.x >> 5);
  int ng = wid % NG;
  int mt = (wid / NG) % MT;
  int sx = wid / (NG * MT);

  int lane  = threadIdx.x & 31;
  int lrow  = lane & 15;
  int khalf = lane >> 4;
  int rowbase = mt * 16;

  const float* arow = x + (size_t)(rowbase + lrow) * DIM;
  const float* sgp[NSUB1];
  const float* sup[NSUB1];
#pragma unroll
  for (int s = 0; s < NSUB1; ++s) {
    int col = (ng * NSUB1 + s) * 16 + lrow;
    sgp[s] = Sg + (size_t)sx * DIM * HSH + col;
    sup[s] = Su + (size_t)sx * DIM * HSH + col;
  }

  v8f accg[NSUB1] = {}, accu[NSUB1] = {};
  for (int k0 = 0; k0 < DIM; k0 += 32) {
    int ka = k0 + khalf * 16;
    v16bf a = load16_cvt(arow + ka);
#pragma unroll
    for (int s = 0; s < NSUB1; ++s) {
      v16bf bg = load16_cvt_strided(sgp[s] + (size_t)ka * HSH, HSH);
      accg[s] = wmma_bf16(a, bg, accg[s]);
      v16bf bu = load16_cvt_strided(sup[s] + (size_t)ka * HSH, HSH);
      accu[s] = wmma_bf16(a, bu, accu[s]);
    }
  }

#pragma unroll
  for (int v = 0; v < 8; ++v) {
    int row = rowbase + v + khalf * 8;
    bf16_t* hr = hsbuf + ((size_t)sx * N_TOK + row) * HSH;
#pragma unroll
    for (int s = 0; s < NSUB1; ++s) {
      int ncol = (ng * NSUB1 + s) * 16 + lrow;
      hr[ncol] = (bf16_t)(silu(accg[s][v]) * accu[s][v]);
    }
  }
}

// ---------------------------------------------------------------------------
// 6) shared experts GEMM2: out += sum_s hs[s] @ Sd[s]. Unique (row,col)
//    ownership -> plain RMW; runs after expert combine on the stream.
// ---------------------------------------------------------------------------
__global__ void moe_shared_gemm2(const bf16_t* __restrict__ hsbuf,
                                 const float* __restrict__ Sd,
                                 float* __restrict__ out) {
  constexpr int NG = (DIM / 16) / NSUB2;  // 32
  int wid = blockIdx.x * (blockDim.x >> 5) + (threadIdx.x >> 5);
  int ng = wid % NG;
  int mt = wid / NG;

  int lane  = threadIdx.x & 31;
  int lrow  = lane & 15;
  int khalf = lane >> 4;
  int rowbase = mt * 16;

  v8f acc[NSUB2] = {};
  for (int sx = 0; sx < NSH; ++sx) {
    const bf16_t* hrow = hsbuf + ((size_t)sx * N_TOK + rowbase + lrow) * HSH;
    const float* sdp[NSUB2];
#pragma unroll
    for (int s = 0; s < NSUB2; ++s)
      sdp[s] = Sd + (size_t)sx * HSH * DIM + (ng * NSUB2 + s) * 16 + lrow;
    for (int k0 = 0; k0 < HSH; k0 += 32) {
      int ka = k0 + khalf * 16;
      v16bf a = *(const v16bf*)(hrow + ka);
#pragma unroll
      for (int s = 0; s < NSUB2; ++s) {
        v16bf b = load16_cvt_strided(sdp[s] + (size_t)ka * DIM, DIM);
        acc[s] = wmma_bf16(a, b, acc[s]);
      }
    }
  }

#pragma unroll
  for (int v = 0; v < 8; ++v) {
    int row = rowbase + v + khalf * 8;
    float* orow = out + (size_t)row * DIM;
#pragma unroll
    for (int s = 0; s < NSUB2; ++s) {
      int ncol = (ng * NSUB2 + s) * 16 + lrow;
      orow[ncol] += acc[s][v];
    }
  }
}

// ---------------------------------------------------------------------------
// launch
// ---------------------------------------------------------------------------
extern "C" void kernel_launch(void* const* d_in, const int* in_sizes, int n_in,
                              void* d_out, int out_size, void* d_ws, size_t ws_size,
                              hipStream_t stream) {
  (void)in_sizes; (void)n_in; (void)out_size; (void)ws_size;
  const float* x    = (const float*)d_in[0];
  const float* Wr   = (const float*)d_in[1];
  const float* bias = (const float*)d_in[2];
  const float* Wg   = (const float*)d_in[3];
  const float* Wu   = (const float*)d_in[4];
  const float* Wd   = (const float*)d_in[5];
  const float* Sg   = (const float*)d_in[6];
  const float* Su   = (const float*)d_in[7];
  const float* Sd   = (const float*)d_in[8];
  float* out = (float*)d_out;

  // workspace carve-out (~51 MB), 256B aligned chunks
  char*  ws  = (char*)d_ws;
  size_t off = 0;
  auto carve = [&](size_t bytes) -> char* {
    char* p = ws + off;
    off = (off + bytes + 255) & ~(size_t)255;
    return p;
  };
  int*    eid      = (int*)   carve((size_t)N_TOK * TOPK * sizeof(int));
  float*  wts      = (float*) carve((size_t)N_TOK * TOPK * sizeof(float));
  int*    counts   = (int*)   carve((size_t)NEXP * sizeof(int));
  int*    slot_tok = (int*)   carve((size_t)NEXP * CAP * sizeof(int));
  float*  slot_w   = (float*) carve((size_t)NEXP * CAP * sizeof(float));
  bf16_t* hbuf     = (bf16_t*)carve((size_t)NEXP * CAP * HDIM * sizeof(bf16_t));
  bf16_t* hsbuf    = (bf16_t*)carve((size_t)NSH * N_TOK * HSH * sizeof(bf16_t));

  moe_zero_out<<<(N_TOK * DIM / 4) / 256, 256, 0, stream>>>(out);
  moe_router<<<N_TOK / 8, 256, 0, stream>>>(x, Wr, bias, eid, wts);
  moe_build_slots<<<1, 64, 0, stream>>>(eid, wts, counts, slot_tok, slot_w);

  // waves = E * MT * NG; blocks = waves / 8
  moe_expert_gemm1<<<(NEXP * (CAP/16) * ((HDIM/16)/NSUB1)) / 8, 256, 0, stream>>>(
      x, Wg, Wu, counts, slot_tok, hbuf);
  moe_expert_gemm2<<<(NEXP * (CAP/16) * ((DIM/16)/NSUB2)) / 8, 256, 0, stream>>>(
      hbuf, Wd, counts, slot_tok, slot_w, out);
  moe_shared_gemm1<<<(NSH * (N_TOK/16) * ((HSH/16)/NSUB1)) / 8, 256, 0, stream>>>(
      x, Sg, Su, hsbuf);
  moe_shared_gemm2<<<((N_TOK/16) * ((DIM/16)/NSUB2)) / 8, 256, 0, stream>>>(
      hsbuf, Sd, out);
}
